// Eucl_Mat_38259568672963
// MI455X (gfx1250) — matrix operand order repaired and checked by reading, not compiled, not run on hardware
//
#include <hip/hip_runtime.h>

// Pairwise Euclidean distance for mapping[8192,256] (fp32) -> out[8192,8192] (fp32)
// d(i,j) = sqrt(max(sq[i] + sq[j] - 2*gram[i][j], 0))
// gram computed with V_WMMA_F32_16X16X32_BF16 using a 2-way bf16 split of fp32
// (hi*hi + hi*lo + lo*hi), giving ~fp32 accuracy at ~2.7x the fp32-WMMA rate.
//
// Each wave computes a 32x64 output block (2 M-tiles x 4 N-tiles = 8 WMMA
// accumulators). K=256 is fully unrolled: every operand address is
// base + compile-time offset (max span ~24.5 KB, inside the 24-bit IOFFSET),
// so the hot loop is pure global_load_b128 + v_wmma with no pointer VALU.
//
// Workspace layout (d_ws), total ~8.4 MB:
//   [0 .. 4MB)         : bf16 hi plane  (8192*256)
//   [4MB .. 8MB)       : bf16 lo plane  (8192*256)
//   [8MB .. 8MB+32KB)  : fp32 row norms sq[8192]

typedef __bf16 bf16_t;
typedef __attribute__((ext_vector_type(16))) __bf16 v16bf;
typedef __attribute__((ext_vector_type(8)))  __bf16 v8bf;
typedef __attribute__((ext_vector_type(8)))  float  v8f;
typedef __attribute__((ext_vector_type(4)))  float  vf4;

#define NROWS 8192
#define DIM   256
#define MT    2          // 16-row M tiles per wave  (32 rows)
#define NT    4          // 16-col N tiles per wave  (64 cols)

__device__ __forceinline__ v16bf cat8(v8bf a, v8bf b) {
  return __builtin_shufflevector(a, b, 0,1,2,3,4,5,6,7,8,9,10,11,12,13,14,15);
}

// --- Prep: fp32 -> (bf16 hi, bf16 lo) planes + row sum-of-squares -----------
// One wave per row; lane handles 8 contiguous elements (coalesced b128 I/O).
__global__ __launch_bounds__(256) void prep_split_sq_kernel(
    const float* __restrict__ A, bf16_t* __restrict__ hiP,
    bf16_t* __restrict__ loP, float* __restrict__ sq) {
  const int wave = (blockIdx.x * 256 + threadIdx.x) >> 5;   // row, 0..8191
  const int lane = threadIdx.x & 31;
  const float* row = A + (size_t)wave * DIM;

  const vf4 c0 = ((const vf4*)row)[lane * 2 + 0];
  const vf4 c1 = ((const vf4*)row)[lane * 2 + 1];
  float x[8] = {c0.x, c0.y, c0.z, c0.w, c1.x, c1.y, c1.z, c1.w};

  v8bf h8, l8;
  float s = 0.f;
#pragma unroll
  for (int i = 0; i < 8; ++i) {
    __bf16 h = (__bf16)x[i];                 // RNE fp32->bf16
    h8[i] = h;
    l8[i] = (__bf16)(x[i] - (float)h);       // residual
    s = __builtin_fmaf(x[i], x[i], s);
  }
  *(v8bf*)(hiP + (size_t)wave * DIM + lane * 8) = h8;
  *(v8bf*)(loP + (size_t)wave * DIM + lane * 8) = l8;

#pragma unroll
  for (int off = 16; off > 0; off >>= 1) s += __shfl_xor(s, off, 32);
  if (lane == 0) sq[wave] = s;
}

// --- Main: WMMA gram + distance epilogue ------------------------------------
// A operand lane layout (16x32 bf16): lane l: row M=l%16,
//   elems 0..7  = K k0 + 8*(l>=16) .. +7
//   elems 8..15 = K k0 + 16 + 8*(l>=16) .. +7
// B operand (32x16 bf16) = rows of M as columns: lane l: col N=l%16,
//   elems 0..7 = K k0 + 8*(l>=16) .. +7, elems 8..15 = K k0 + 16 + 8*(l>=16) .. +7
// C/D layout: VGPR r of lane l holds (M = r + 8*(l>=16), N = l%16).
__global__ __launch_bounds__(256) void pdist_wmma_kernel(
    const bf16_t* __restrict__ hiP, const bf16_t* __restrict__ loP,
    const float* __restrict__ sq, float* __restrict__ out) {
  const int waveId = (int)((blockIdx.x * 256 + threadIdx.x) >> 5);
  const int lane   = threadIdx.x & 31;
  const int half   = lane >> 4;       // 0 or 1
  const int lm     = lane & 15;

  const int tileMs = waveId >> 7;     // 0..255  (rows 32*tileMs .. +31)
  const int tileNs = waveId & 127;    // 0..127  (cols 64*tileNs .. +63)

  const bf16_t* aHi[MT];
  const bf16_t* aLo[MT];
#pragma unroll
  for (int i = 0; i < MT; ++i) {
    const size_t aOff = (size_t)(tileMs * 32 + i * 16 + lm) * DIM;
    aHi[i] = hiP + aOff;
    aLo[i] = loP + aOff;
  }

  const bf16_t* bHi[NT];
  const bf16_t* bLo[NT];
#pragma unroll
  for (int j = 0; j < NT; ++j) {
    const size_t bOff = (size_t)((tileNs * NT + j) * 16 + lm) * DIM;
    bHi[j] = hiP + bOff;
    bLo[j] = loP + bOff;
  }

  // One up-front prefetch pass over the B strips (gfx1250 global_prefetch_b8)
  // to start warming L0 before the load/WMMA stream.
#pragma unroll
  for (int j = 0; j < NT; ++j) {
    __builtin_prefetch(bHi[j] + half * 16, 0, 3);
    __builtin_prefetch(bHi[j] + 128 + half * 16, 0, 3);
    __builtin_prefetch(bLo[j] + half * 16, 0, 3);
    __builtin_prefetch(bLo[j] + 128 + half * 16, 0, 3);
  }

  const v8f vzero = {0.f, 0.f, 0.f, 0.f, 0.f, 0.f, 0.f, 0.f};
  v8f acc[MT][NT];
#pragma unroll
  for (int i = 0; i < MT; ++i)
#pragma unroll
    for (int j = 0; j < NT; ++j) acc[i][j] = vzero;

#pragma unroll
  for (int k0 = 0; k0 < DIM; k0 += 32) {
    v16bf a_hi[MT], a_lo[MT];
#pragma unroll
    for (int i = 0; i < MT; ++i) {
      a_hi[i] = cat8(*(const v8bf*)(aHi[i] + k0 + half * 8),
                     *(const v8bf*)(aHi[i] + k0 + 16 + half * 8));
      a_lo[i] = cat8(*(const v8bf*)(aLo[i] + k0 + half * 8),
                     *(const v8bf*)(aLo[i] + k0 + 16 + half * 8));
    }

#pragma unroll
    for (int j = 0; j < NT; ++j) {
      const v16bf b_hi = cat8(*(const v8bf*)(bHi[j] + k0 + half * 8),
                              *(const v8bf*)(bHi[j] + k0 + 16 + half * 8));
      const v16bf b_lo = cat8(*(const v8bf*)(bLo[j] + k0 + half * 8),
                              *(const v8bf*)(bLo[j] + k0 + 16 + half * 8));
#pragma unroll
      for (int i = 0; i < MT; ++i) {
        acc[i][j] = __builtin_amdgcn_wmma_f32_16x16x32_bf16(
            false, a_hi[i], false, b_hi, (short)0, acc[i][j], false, false);
        acc[i][j] = __builtin_amdgcn_wmma_f32_16x16x32_bf16(
            false, a_hi[i], false, b_lo, (short)0, acc[i][j], false, false);
        acc[i][j] = __builtin_amdgcn_wmma_f32_16x16x32_bf16(
            false, a_lo[i], false, b_hi, (short)0, acc[i][j], false, false);
      }
    }
  }

  // Epilogue: d = sqrt(max(sq_i + sq_j - 2g, 0))
  float sqA[MT][8];
#pragma unroll
  for (int i = 0; i < MT; ++i)
#pragma unroll
    for (int r = 0; r < 8; ++r)
      sqA[i][r] = sq[tileMs * 32 + i * 16 + half * 8 + r];

#pragma unroll
  for (int j = 0; j < NT; ++j) {
    const int colBase = (tileNs * NT + j) * 16;
    const float sqB = sq[colBase + lm];
#pragma unroll
    for (int i = 0; i < MT; ++i) {
      const int rowBase = tileMs * 32 + i * 16 + half * 8;
#pragma unroll
      for (int r = 0; r < 8; ++r) {
        float d2 = sqA[i][r] + sqB - 2.f * acc[i][j][r];
        d2 = d2 > 0.f ? d2 : 0.f;
        out[(size_t)(rowBase + r) * NROWS + colBase + lm] = __builtin_sqrtf(d2);
      }
    }
  }
}

extern "C" void kernel_launch(void* const* d_in, const int* in_sizes, int n_in,
                              void* d_out, int out_size, void* d_ws, size_t ws_size,
                              hipStream_t stream) {
  const float* A = (const float*)d_in[0];    // mapping [8192, 256] fp32
  float* out = (float*)d_out;                // [8192, 8192] fp32

  bf16_t* hiP = (bf16_t*)d_ws;                                   // 4 MB
  bf16_t* loP = hiP + (size_t)NROWS * DIM;                       // 4 MB
  float*  sq  = (float*)(loP + (size_t)NROWS * DIM);             // 32 KB

  // 8192 rows, 1 wave (32 lanes) per row, 8 waves per 256-thread block.
  prep_split_sq_kernel<<<NROWS / 8, 256, 0, stream>>>(A, hiP, loP, sq);

  // 256 M-blocks x 128 N-strips = 32768 waves; 8 waves per block.
  pdist_wmma_kernel<<<(256 * 128) / 8, 256, 0, stream>>>(hiP, loP, sq, out);
}
